// StressGCN_UNet_28724741275675
// MI455X (gfx1250) — compile-verified, matrix operand order checked by reading, not run on hardware
//
#include <hip/hip_runtime.h>

// ---------------------------------------------------------------------------
// GraphUNet (StressGCN) for gfx1250 (MI455X).
// Dominant cost: dense B@B (augment) and A_norm@(H@W) GEMMs -> bf16 WMMA,
// fp32 accumulate, everything L2-resident (192MB >> 130MB working set).
// ---------------------------------------------------------------------------

typedef __attribute__((ext_vector_type(16))) __bf16          v16bf;
typedef __attribute__((ext_vector_type(8)))  float           v8f;
typedef __attribute__((ext_vector_type(8)))  unsigned short  ushort8;

union BfFrag { ushort8 u[2]; v16bf v; };

static __device__ __forceinline__ unsigned short f2bf(float f) {
    union { float f; unsigned u; } v; v.f = f;
    unsigned r = v.u + 0x7FFFu + ((v.u >> 16) & 1u);   // round-to-nearest-even
    return (unsigned short)(r >> 16);
}

// ---------------------------------------------------------------------------
// Core WMMA GEMM: C[M,N] (f32, row-major) = A[M,K](bf16,row-major) * B,
// where B is supplied K-major as Bt[N,K] (i.e. B column-major).
// Block: 256 threads = 8 waves, 64x128 C-tile; wave = 32x32 via 4 accumulators.
// Requires M%64==0, N%128==0, K%32==0 (true for all call sites here).
// ---------------------------------------------------------------------------
__global__ __launch_bounds__(256)
void gunet_gemm_bf16_wmma(const unsigned short* __restrict__ A,
                          const unsigned short* __restrict__ Bt,
                          float* __restrict__ C,
                          int M, int N, int K,
                          const float* __restrict__ bias, int relu)
{
    const int lane  = threadIdx.x & 31;
    const int wave  = threadIdx.x >> 5;
    const int waveR = wave >> 2;            // 0..1
    const int waveC = wave & 3;             // 0..3
    const int tileM = blockIdx.x * 64  + waveR * 32;
    const int tileN = blockIdx.y * 128 + waveC * 32;
    const int lr    = lane & 15;
    const int hsel  = lane >> 4;            // which K-half this lane holds

    v8f acc00 = {}, acc01 = {}, acc10 = {}, acc11 = {};

    // Per CDNA5 ISA 16-bit A layout: lane half h holds K = [8h..8h+7] U [16+8h..16+8h+7]
    const unsigned short* arow0 = A  + (size_t)(tileM + lr) * K + 8 * hsel;
    const unsigned short* arow1 = arow0 + (size_t)16 * K;
    const unsigned short* brow0 = Bt + (size_t)(tileN + lr) * K + 8 * hsel;
    const unsigned short* brow1 = brow0 + (size_t)16 * K;

    for (int k0 = 0; k0 < K; k0 += 32) {
        __builtin_prefetch(arow0 + k0 + 128, 0, 0);   // global_prefetch_b8 (speculative)
        __builtin_prefetch(brow0 + k0 + 128, 0, 0);

        BfFrag a0, a1, b0, b1;
        a0.u[0] = *(const ushort8*)(arow0 + k0);
        a0.u[1] = *(const ushort8*)(arow0 + k0 + 16);
        a1.u[0] = *(const ushort8*)(arow1 + k0);
        a1.u[1] = *(const ushort8*)(arow1 + k0 + 16);
        b0.u[0] = *(const ushort8*)(brow0 + k0);
        b0.u[1] = *(const ushort8*)(brow0 + k0 + 16);
        b1.u[0] = *(const ushort8*)(brow1 + k0);
        b1.u[1] = *(const ushort8*)(brow1 + k0 + 16);

        acc00 = __builtin_amdgcn_wmma_f32_16x16x32_bf16(false, a0.v, false, b0.v,
                                                        (short)0, acc00, false, false);
        acc01 = __builtin_amdgcn_wmma_f32_16x16x32_bf16(false, a0.v, false, b1.v,
                                                        (short)0, acc01, false, false);
        acc10 = __builtin_amdgcn_wmma_f32_16x16x32_bf16(false, a1.v, false, b0.v,
                                                        (short)0, acc10, false, false);
        acc11 = __builtin_amdgcn_wmma_f32_16x16x32_bf16(false, a1.v, false, b1.v,
                                                        (short)0, acc11, false, false);
    }

    // C/D layout: VGPR r -> M=r (lanes 0-15) / M=8+r (lanes 16-31), N = lane&15.
    const int c0 = tileN + lr;
    const int c1 = c0 + 16;
#pragma unroll
    for (int r = 0; r < 8; ++r) {
        int row0 = tileM + r + hsel * 8;
        int row1 = row0 + 16;
        float v;
        v = acc00[r]; if (bias) v += bias[c0]; if (relu) v = fmaxf(v, 0.f);
        C[(size_t)row0 * N + c0] = v;
        v = acc01[r]; if (bias) v += bias[c1]; if (relu) v = fmaxf(v, 0.f);
        C[(size_t)row0 * N + c1] = v;
        v = acc10[r]; if (bias) v += bias[c0]; if (relu) v = fmaxf(v, 0.f);
        C[(size_t)row1 * N + c0] = v;
        v = acc11[r]; if (bias) v += bias[c1]; if (relu) v = fmaxf(v, 0.f);
        C[(size_t)row1 * N + c1] = v;
    }
}

// ---------------------------------------------------------------------------
// Support kernels
// ---------------------------------------------------------------------------
__global__ void gunet_scatter_edges(const int* __restrict__ ei, float* __restrict__ A,
                                    int n, int ne)
{
    int e = blockIdx.x * blockDim.x + threadIdx.x;
    if (e < ne) atomicAdd(&A[(size_t)ei[ne + e] * n + ei[e]], 1.0f);
}

__global__ void gunet_encode(const float* __restrict__ x, const float* __restrict__ W,
                             const float* __restrict__ b, float* __restrict__ h)
{
    int i = blockIdx.x, c = threadIdx.x;            // 4096 x 128
    float s = b[c];
#pragma unroll
    for (int k = 0; k < 16; ++k) s += x[i * 16 + k] * W[k * 128 + c];
    h[(size_t)i * 128 + c] = s;
}

__global__ void gunet_rowsum_dis(const float* __restrict__ A, float* __restrict__ dis, int n)
{
    __shared__ float red[256];
    int i = blockIdx.x;
    float s = 0.f;
    for (int j = threadIdx.x; j < n; j += 256) s += A[(size_t)i * n + j];
    red[threadIdx.x] = s; __syncthreads();
    for (int st = 128; st > 0; st >>= 1) {
        if (threadIdx.x < st) red[threadIdx.x] += red[threadIdx.x + st];
        __syncthreads();
    }
    if (threadIdx.x == 0) dis[i] = rsqrtf(red[0] + 2.0f);   // A_hat = A + 2I
}

__global__ void gunet_anorm_cvt(const float* __restrict__ A, const float* __restrict__ dis,
                                unsigned short* __restrict__ Anb, int n)
{
    size_t idx = (size_t)blockIdx.x * blockDim.x + threadIdx.x;
    size_t tot = (size_t)n * n;
    if (idx >= tot) return;
    int i = (int)(idx / (size_t)n), j = (int)(idx % (size_t)n);
    float v = A[idx] + (i == j ? 2.0f : 0.0f);
    Anb[idx] = f2bf(v * dis[i] * dis[j]);
}

__global__ void gunet_cvt_bf16(const float* __restrict__ in, unsigned short* __restrict__ out,
                               size_t nElem)
{
    size_t i = (size_t)blockIdx.x * blockDim.x + threadIdx.x;
    if (i < nElem) out[i] = f2bf(in[i]);
}

// f32 [R x C] -> bf16 [C x R] (LDS-tiled), R%32==0, C%32==0 at all call sites
__global__ void gunet_transpose_cvt(const float* __restrict__ in, unsigned short* __restrict__ out,
                                    int R, int C)
{
    __shared__ float tile[32][33];
    int bx = blockIdx.x * 32;   // along C
    int by = blockIdx.y * 32;   // along R
    for (int dy = threadIdx.y; dy < 32; dy += 8)
        tile[dy][threadIdx.x] = in[(size_t)(by + dy) * C + (bx + threadIdx.x)];
    __syncthreads();
    for (int dy = threadIdx.y; dy < 32; dy += 8)
        out[(size_t)(bx + dy) * R + (by + threadIdx.x)] = f2bf(tile[threadIdx.x][dy]);
}

// B = A with diag forced to 1; write B (bf16 row-major) and B^T (bf16 row-major)
__global__ void gunet_make_B_Bt(const float* __restrict__ A, unsigned short* __restrict__ Bb,
                                unsigned short* __restrict__ Btb, int n)
{
    __shared__ float tile[32][33];
    int bx = blockIdx.x * 32, by = blockIdx.y * 32;
    for (int dy = threadIdx.y; dy < 32; dy += 8) {
        int r = by + dy, c = bx + threadIdx.x;
        float v = A[(size_t)r * n + c];
        if (r == c) v = 1.0f;
        tile[dy][threadIdx.x] = v;
        Bb[(size_t)r * n + c] = f2bf(v);
    }
    __syncthreads();
    for (int dy = threadIdx.y; dy < 32; dy += 8)
        Btb[(size_t)(bx + dy) * n + (by + threadIdx.x)] = f2bf(tile[threadIdx.x][dy]);
}

__global__ void gunet_zero_diag(float* __restrict__ Cg, int n)
{
    int i = blockIdx.x * blockDim.x + threadIdx.x;
    if (i < n) Cg[(size_t)i * n + i] = 0.f;
}

__global__ void gunet_pnorm(const float* __restrict__ p, float* __restrict__ out)
{
    __shared__ float red[128];
    float v = p[threadIdx.x];
    red[threadIdx.x] = v * v; __syncthreads();
    for (int st = 64; st > 0; st >>= 1) {
        if (threadIdx.x < st) red[threadIdx.x] += red[threadIdx.x + st];
        __syncthreads();
    }
    if (threadIdx.x == 0) out[0] = sqrtf(red[0]);
}

__global__ void gunet_score(const float* __restrict__ h, const float* __restrict__ p,
                            const float* __restrict__ pn, float* __restrict__ score, int n)
{
    int i = blockIdx.x * blockDim.x + threadIdx.x;
    if (i >= n) return;
    float s = 0.f;
#pragma unroll 8
    for (int c = 0; c < 128; ++c) s += h[(size_t)i * 128 + c] * p[c];
    score[i] = s / pn[0];
}

// Stable descending rank-select == jax.lax.top_k ordering (ties -> lower index)
__global__ void gunet_rank_select(const float* __restrict__ score, int* __restrict__ perm,
                                  int n, int k)
{
    int i = blockIdx.x * blockDim.x + threadIdx.x;
    if (i >= n) return;
    float si = score[i];
    int rank = 0;
    for (int j = 0; j < n; ++j) {
        float sj = score[j];
        rank += (sj > si) || (sj == si && j < i);
    }
    if (rank < k) perm[rank] = i;
}

__global__ void gunet_pool_h(const float* __restrict__ h, const float* __restrict__ score,
                             const int* __restrict__ perm, float* __restrict__ hp, int k)
{
    int r = blockIdx.x, c = threadIdx.x;
    int src = perm[r];
    hp[(size_t)r * 128 + c] = h[(size_t)src * 128 + c] * tanhf(score[src]);
}

__global__ void gunet_pool_A(const float* __restrict__ Cg, const int* __restrict__ perm,
                             float* __restrict__ Ap, int n, int k)
{
    size_t idx = (size_t)blockIdx.x * blockDim.x + threadIdx.x;
    if (idx >= (size_t)k * k) return;
    int r = (int)(idx / (size_t)k), c = (int)(idx % (size_t)k);
    Ap[idx] = Cg[(size_t)perm[r] * n + perm[c]];
}

__global__ void gunet_copy_f32(const float* __restrict__ in, float* __restrict__ out, size_t nElem)
{
    size_t i = (size_t)blockIdx.x * blockDim.x + threadIdx.x;
    if (i < nElem) out[i] = in[i];
}

__global__ void gunet_unpool_scatter(const float* __restrict__ h, const int* __restrict__ perm,
                                     float* __restrict__ out, int k)
{
    int r = blockIdx.x, c = threadIdx.x;   // perm entries unique -> no races
    out[(size_t)perm[r] * 128 + c] += h[(size_t)r * 128 + c];
}

__global__ void gunet_matvec_hW(const float* __restrict__ h, const float* __restrict__ Wout,
                                float* __restrict__ hv, int n)
{
    int i = blockIdx.x * blockDim.x + threadIdx.x;
    if (i >= n) return;
    float s = 0.f;
#pragma unroll 8
    for (int c = 0; c < 128; ++c) s += h[(size_t)i * 128 + c] * Wout[c];
    hv[i] = s;
}

__global__ void gunet_final_conv(const float* __restrict__ A, const float* __restrict__ dis,
                                 const float* __restrict__ hv, const float* __restrict__ bout,
                                 float* __restrict__ out, int n)
{
    __shared__ float red[256];
    int i = blockIdx.x;
    float s = 0.f;
    for (int j = threadIdx.x; j < n; j += 256) {
        float a = A[(size_t)i * n + j] + (j == i ? 2.0f : 0.0f);
        s += a * dis[j] * hv[j];
    }
    red[threadIdx.x] = s; __syncthreads();
    for (int st = 128; st > 0; st >>= 1) {
        if (threadIdx.x < st) red[threadIdx.x] += red[threadIdx.x + st];
        __syncthreads();
    }
    if (threadIdx.x == 0) out[i] = dis[i] * red[0] + bout[0];
}

// ---------------------------------------------------------------------------
// Host orchestration
// ---------------------------------------------------------------------------
extern "C" void kernel_launch(void* const* d_in, const int* in_sizes, int n_in,
                              void* d_out, int out_size, void* d_ws, size_t ws_size,
                              hipStream_t stream)
{
    (void)in_sizes; (void)n_in; (void)out_size; (void)ws_size;

    const float* x     = (const float*)d_in[0];
    const int*   ei    = (const int*)  d_in[1];   // [2, 65536], row1=dst, row0=src
    /* d_in[2] = batch (unused, single graph) */
    const float* Wenc  = (const float*)d_in[3];
    const float* benc  = (const float*)d_in[4];
    const float* Wdown = (const float*)d_in[5];   // [5,128,128]
    const float* bdown = (const float*)d_in[6];   // [5,128]
    const float* Wup   = (const float*)d_in[7];   // [3,128,128]
    const float* bup   = (const float*)d_in[8];   // [3,128]
    const float* Wout  = (const float*)d_in[9];   // [128,1]
    const float* bout  = (const float*)d_in[10];  // [1]
    const float* ppool = (const float*)d_in[11];  // [4,128]
    float* out = (float*)d_out;                   // [4096,1] f32

    static const int ns[5] = {4096, 2048, 1024, 512, 256};
    const int NE = 65536;

    // --- bump allocator over d_ws ---
    char* base = (char*)d_ws;
    size_t off = 0;
    auto alloc = [&](size_t bytes) -> void* {
        void* p = base + off;
        off = (off + bytes + 255) & ~(size_t)255;
        return p;
    };

    float*          A[5];  unsigned short* Anb[5];  float* dis[5];
    for (int i = 0; i < 5; ++i) A[i]   = (float*)alloc((size_t)ns[i] * ns[i] * 4);
    for (int i = 0; i < 5; ++i) Anb[i] = (unsigned short*)alloc((size_t)ns[i] * ns[i] * 2);
    for (int i = 0; i < 5; ++i) dis[i] = (float*)alloc((size_t)ns[i] * 4);
    float*          Caug = (float*)alloc((size_t)4096 * 4096 * 4);
    unsigned short* Bbf  = (unsigned short*)alloc((size_t)4096 * 4096 * 2);
    unsigned short* Btbf = (unsigned short*)alloc((size_t)4096 * 4096 * 2);
    float* xs[4];
    for (int i = 0; i < 4; ++i) xs[i] = (float*)alloc((size_t)ns[i] * 128 * 4);
    int* perm[4];
    for (int i = 0; i < 4; ++i) perm[i] = (int*)alloc((size_t)ns[i + 1] * 4);
    float*          henc  = (float*)alloc((size_t)4096 * 128 * 4);
    float*          hpool = (float*)alloc((size_t)2048 * 128 * 4);
    float*          hd4   = (float*)alloc((size_t)256  * 128 * 4);
    float*          hua   = (float*)alloc((size_t)4096 * 128 * 4);
    float*          hsum  = (float*)alloc((size_t)4096 * 128 * 4);
    float*          t1    = (float*)alloc((size_t)4096 * 128 * 4);
    unsigned short* hbf   = (unsigned short*)alloc((size_t)4096 * 128 * 2);
    unsigned short* Wt    = (unsigned short*)alloc((size_t)128 * 128 * 2);
    unsigned short* C1t   = (unsigned short*)alloc((size_t)128 * 4096 * 2);
    float*          score = (float*)alloc((size_t)4096 * 4);
    float*          pn    = (float*)alloc(256);
    float*          hv    = (float*)alloc((size_t)4096 * 4);

    // --- helpers (host lambdas, all launches on `stream`) ---
    auto anorm_prep = [&](int n, const float* Af, float* disb, unsigned short* Anorm) {
        gunet_rowsum_dis<<<n, 256, 0, stream>>>(Af, disb, n);
        size_t tot = (size_t)n * n;
        gunet_anorm_cvt<<<(unsigned)((tot + 255) / 256), 256, 0, stream>>>(Af, disb, Anorm, n);
    };

    // GCN conv: hout = relu?( A_norm @ (hin @ W) + b )
    auto conv = [&](int n, const unsigned short* Anorm, const float* hin,
                    const float* W, const float* b, int relu, float* hout) {
        gunet_cvt_bf16<<<(unsigned)(((size_t)n * 128 + 255) / 256), 256, 0, stream>>>(hin, hbf, (size_t)n * 128);
        gunet_transpose_cvt<<<dim3(4, 4), dim3(32, 8), 0, stream>>>(W, Wt, 128, 128);
        gunet_gemm_bf16_wmma<<<dim3(n / 64, 1), 256, 0, stream>>>(hbf, Wt, t1, n, 128, 128, nullptr, 0);
        gunet_transpose_cvt<<<dim3(4, n / 32), dim3(32, 8), 0, stream>>>(t1, C1t, n, 128);
        gunet_gemm_bf16_wmma<<<dim3(n / 64, 1), 256, 0, stream>>>(Anorm, C1t, hout, n, 128, n, b, relu);
    };

    auto augment = [&](int n, const float* Af) {
        gunet_make_B_Bt<<<dim3(n / 32, n / 32), dim3(32, 8), 0, stream>>>(Af, Bbf, Btbf, n);
        gunet_gemm_bf16_wmma<<<dim3(n / 64, n / 128), 256, 0, stream>>>(Bbf, Btbf, Caug, n, n, n, nullptr, 0);
        gunet_zero_diag<<<(n + 255) / 256, 256, 0, stream>>>(Caug, n);
    };

    auto pool = [&](int n, int k, const float* h, const float* p, int* pm,
                    float* hp, float* Anext) {
        gunet_pnorm<<<1, 128, 0, stream>>>(p, pn);
        gunet_score<<<(n + 255) / 256, 256, 0, stream>>>(h, p, pn, score, n);
        gunet_rank_select<<<(n + 255) / 256, 256, 0, stream>>>(score, pm, n, k);
        gunet_pool_h<<<k, 128, 0, stream>>>(h, score, pm, hp, k);
        size_t kk = (size_t)k * k;
        gunet_pool_A<<<(unsigned)((kk + 255) / 256), 256, 0, stream>>>(Caug, pm, Anext, n, k);
    };

    // --- build adjacency + encoder ---
    hipMemsetAsync(A[0], 0, (size_t)4096 * 4096 * 4, stream);
    gunet_scatter_edges<<<(NE + 255) / 256, 256, 0, stream>>>(ei, A[0], 4096, NE);
    gunet_encode<<<4096, 128, 0, stream>>>(x, Wenc, benc, henc);

    // --- down path ---
    anorm_prep(4096, A[0], dis[0], Anb[0]);
    conv(4096, Anb[0], henc, Wdown, bdown, 1, xs[0]);
    const float* hprev = xs[0];
    for (int i = 1; i <= 4; ++i) {
        int np = ns[i - 1], k = ns[i];
        augment(np, A[i - 1]);
        pool(np, k, hprev, ppool + (size_t)(i - 1) * 128, perm[i - 1], hpool, A[i]);
        anorm_prep(k, A[i], dis[i], Anb[i]);
        float* outp = (i < 4) ? xs[i] : hd4;
        conv(k, Anb[i], hpool, Wdown + (size_t)i * 128 * 128, bdown + (size_t)i * 128, 1, outp);
        hprev = outp;
    }

    // --- up path ---
    for (int i = 0; i < 4; ++i) {
        int j  = 3 - i;
        int nj = ns[j];       // target rows (size of xs[j])
        int kk = ns[j + 1];   // current h rows
        gunet_copy_f32<<<(unsigned)(((size_t)nj * 128 + 255) / 256), 256, 0, stream>>>(
            xs[j], hsum, (size_t)nj * 128);
        gunet_unpool_scatter<<<kk, 128, 0, stream>>>(hprev, perm[j], hsum, kk);
        if (i < 3) {
            conv(nj, Anb[j], hsum, Wup + (size_t)i * 128 * 128, bup + (size_t)i * 128, 1, hua);
            hprev = hua;
        } else {
            // final: A_norm @ (h @ W_out[128x1]) + b_out  -> d_out [4096]
            gunet_matvec_hW<<<(4096 + 255) / 256, 256, 0, stream>>>(hsum, Wout, hv, 4096);
            gunet_final_conv<<<4096, 256, 0, stream>>>(A[0], dis[0], hv, bout, out, 4096);
        }
    }
}
